// Net_24154896073047
// MI455X (gfx1250) — compile-verified, matrix operand order checked by reading, not run on hardware
//
#include <hip/hip_runtime.h>
#include <hip/hip_bf16.h>

typedef __attribute__((ext_vector_type(16))) _Float16 v16h;
typedef __attribute__((ext_vector_type(8)))  _Float16 v8h;
typedef __attribute__((ext_vector_type(8)))  float    v8f;

#define T_STEPS 50
#define Bz      128
#define STIM    784
#define S_SIZE  1024
#define R_SIZE  2048
#define NBLK    64
#define TPB     256          // 8 waves/block * 64 blocks = 512 waves total

// ---------------- workspace layout (bytes) ----------------
#define OFF_MEM_S    0u
#define OFF_MEM_R    (OFF_MEM_S + Bz*S_SIZE*4u)            // 524288
#define OFF_MEM_O    (OFF_MEM_R + Bz*R_SIZE*4u)            // 1572864
#define OFF_SSPK     (OFF_MEM_O + Bz*16*4u)                // 1581056  [2][128][1024] f16
#define OFF_RSPK     (OFF_SSPK + 2u*Bz*S_SIZE*2u)          // 2105344  [2][128][2048] f16
#define OFF_OSPK     (OFF_RSPK + 2u*Bz*R_SIZE*2u)          // 3153920  [2][128][32]   f16
#define OFF_BAR      (OFF_OSPK + 2u*Bz*32*2u)              // 3170304  barrier cnt/gen
#define OFF_ZERO_END (OFF_BAR + 256u)                      // everything below zeroed per call
#define OFF_SFF      OFF_ZERO_END                          // [128][1024] f32
#define OFF_WSR16    (OFF_SFF + Bz*S_SIZE*4u)
#define OFF_WRD16    (OFF_WSR16 + S_SIZE*S_SIZE*2u)
#define OFF_WRR16    (OFF_WRD16 + (unsigned)R_SIZE*S_SIZE*2u)
#define OFF_WOD16    (OFF_WRR16 + (unsigned)R_SIZE*R_SIZE*2u)   // [16][2048] f16 (N padded)
#define OFF_WOR16    (OFF_WOD16 + 16u*R_SIZE*2u)                // [16][32] f16 (N,K padded)
#define OFF_BODP     (OFF_WOR16 + 16u*32u*2u)                   // [16] f32 padded
#define OFF_BORP     (OFF_BODP + 64u)                           // [16] f32 padded

// ---------------- WMMA helpers ----------------
__device__ __forceinline__ v8f wmma_f16(v16h a, v16h b, v8f c) {
  // v_wmma_f32_16x16x32_f16  (D = A*B + C, f32 accum)
  return __builtin_amdgcn_wmma_f32_16x16x32_f16(
      /*neg_a=*/false, a, /*neg_b=*/false, b,
      /*c_mod=*/(short)0, c, /*reuse_a=*/false, /*reuse_b=*/false);
}

// Load a 16x32 f16 fragment (A layout; B of an NT GEMM uses the same pattern on
// row-major W[N][K]).  Lane l: row = l&15, k-offset = (l>>4)*8; elements
// [0..7] = K+0..7, [8..15] = K+16..23 per the CDNA5 16-bit A VGPR layout.
// Works for global pointers (global_load_b128) and for pointers derived from
// __shared__ arrays (ds_load_b128 after address-space inference).
__device__ __forceinline__ v16h load_frag(const _Float16* p0, int stride, int lane) {
  const int rr = lane & 15;
  const int ko = (lane >> 4) << 3;
  const _Float16* p = p0 + rr * stride + ko;
  v8h lo = *(const v8h*)(p);
  v8h hi = *(const v8h*)(p + 16);
  v16h v;
#pragma unroll
  for (int i = 0; i < 8; ++i) { v[i] = lo[i]; v[i + 8] = hi[i]; }
  return v;
}

// ---------------- gfx1250 async global->LDS copy ----------------
__device__ __forceinline__ void wait_async0() {
#if __has_builtin(__builtin_amdgcn_s_wait_asynccnt)
  __builtin_amdgcn_s_wait_asynccnt(0);
#else
  asm volatile("s_wait_asynccnt 0x0" ::: "memory");
#endif
}

// Copy 'halfs' contiguous _Float16 from global to LDS using the ASYNCcnt path.
// Generic pointers into __shared__ carry the LDS byte offset in their low 32
// bits (flat-LDS aperture encoding), which is exactly what the instruction's
// LDS-address VGPR wants.  Each thread issues its slice, waits for its own
// asyncs, then the block barrier makes everyone's data visible.
__device__ __forceinline__ void panel_to_lds(const _Float16* __restrict__ g,
                                             _Float16* lds, int halfs) {
  const unsigned lbase = (unsigned)(unsigned long long)lds;
  const int chunks = halfs >> 3;                 // 16B per chunk
  for (int c = threadIdx.x; c < chunks; c += TPB) {
    const unsigned laddr = lbase + (unsigned)(c << 4);
    const unsigned long long gaddr = (unsigned long long)(g + (c << 3));
    asm volatile("global_load_async_to_lds_b128 %0, %1, off"
                 :: "v"(laddr), "v"(gaddr)
                 : "memory");
  }
  wait_async0();
  __syncthreads();
}

// ---------------- device-wide barrier (persistent kernel) ----------------
__device__ __forceinline__ void device_barrier(unsigned* bar) {
  __syncthreads();
  if (threadIdx.x == 0) {
    __threadfence();                      // release spike/mem stores
    unsigned* cnt = bar;
    unsigned* gen = bar + 1;
    unsigned g = __hip_atomic_load(gen, __ATOMIC_RELAXED, __HIP_MEMORY_SCOPE_AGENT);
    unsigned prev = __hip_atomic_fetch_add(cnt, 1u, __ATOMIC_ACQ_REL, __HIP_MEMORY_SCOPE_AGENT);
    if (prev == (unsigned)(NBLK - 1)) {
      __hip_atomic_store(cnt, 0u, __ATOMIC_RELAXED, __HIP_MEMORY_SCOPE_AGENT);
      __hip_atomic_fetch_add(gen, 1u, __ATOMIC_RELEASE, __HIP_MEMORY_SCOPE_AGENT);
    } else {
      while (__hip_atomic_load(gen, __ATOMIC_ACQUIRE, __HIP_MEMORY_SCOPE_AGENT) == g) {
        __builtin_amdgcn_s_sleep(1);
      }
    }
    __threadfence();                      // acquire other blocks' stores
  }
  __syncthreads();
}

// ---------------- prologue kernels ----------------
__global__ void k_zero(unsigned* __restrict__ p, int n) {
  for (int i = blockIdx.x * blockDim.x + threadIdx.x; i < n; i += gridDim.x * blockDim.x)
    p[i] = 0u;
}

__global__ void k_cvt(const float* __restrict__ s, _Float16* __restrict__ d, int n) {
  for (int i = blockIdx.x * blockDim.x + threadIdx.x; i < n; i += gridDim.x * blockDim.x)
    d[i] = (_Float16)s[i];
}

__global__ void k_pad_o(const float* __restrict__ Wod, const float* __restrict__ Wor,
                        const float* __restrict__ bod, const float* __restrict__ bor,
                        char* __restrict__ ws) {
  _Float16* Wod16 = (_Float16*)(ws + OFF_WOD16);   // [16][2048]
  _Float16* Wor16 = (_Float16*)(ws + OFF_WOR16);   // [16][32]
  float* bodp = (float*)(ws + OFF_BODP);
  float* borp = (float*)(ws + OFF_BORP);
  const int total = 16 * R_SIZE + 16 * 32 + 16 + 16;
  for (int i = blockIdx.x * blockDim.x + threadIdx.x; i < total; i += gridDim.x * blockDim.x) {
    if (i < 16 * R_SIZE) {
      int row = i >> 11, col = i & (R_SIZE - 1);
      Wod16[i] = (row < 10) ? (_Float16)Wod[row * R_SIZE + col] : (_Float16)0.0f;
    } else if (i < 16 * R_SIZE + 512) {
      int j = i - 16 * R_SIZE;
      int row = j >> 5, col = j & 31;
      Wor16[j] = (row < 10 && col < 10) ? (_Float16)Wor[row * 10 + col] : (_Float16)0.0f;
    } else if (i < 16 * R_SIZE + 512 + 16) {
      int j = i - (16 * R_SIZE + 512);
      bodp[j] = (j < 10) ? bod[j] : 0.0f;
    } else {
      int j = i - (16 * R_SIZE + 512 + 16);
      borp[j] = (j < 10) ? bor[j] : 0.0f;
    }
  }
}

// one-time exact f32 sensory feed-forward: sff = stimulus @ Wsd.T + bsd
__global__ void k_sff(const float* __restrict__ stim, const float* __restrict__ Wsd,
                      const float* __restrict__ bsd, float* __restrict__ sff) {
  int tid = blockIdx.x * blockDim.x + threadIdx.x;
  if (tid >= Bz * S_SIZE) return;
  int m = tid >> 10, n = tid & (S_SIZE - 1);
  const float* x = stim + m * STIM;
  const float* w = Wsd + n * STIM;
  float acc = bsd[n];
#pragma unroll 4
  for (int k = 0; k < STIM; ++k) acc = fmaf(x[k], w[k], acc);
  sff[tid] = acc;
}

// ---------------- persistent SNN kernel ----------------
__global__ __launch_bounds__(TPB) void snn_persistent(
    const float* __restrict__ noise_s, const float* __restrict__ noise_r,
    const float* __restrict__ noise_o,
    const float* __restrict__ bsr, const float* __restrict__ brd,
    const float* __restrict__ brr,
    char* __restrict__ ws, float* __restrict__ out) {
  // Shared A-panel: all 8 waves of a block work on the same 16-row M-tile, so
  // the spike panel is staged into LDS once per block per GEMM phase.
  __shared__ _Float16 ldsA[16 * R_SIZE];          // 64 KB of the 320 KB WGP pool

  const int lane = threadIdx.x & 31;
  const int wave = (int)((blockIdx.x * TPB + threadIdx.x) >> 5);   // 0..511

  float*    mem_s = (float*)(ws + OFF_MEM_S);
  float*    mem_r = (float*)(ws + OFF_MEM_R);
  float*    mem_o = (float*)(ws + OFF_MEM_O);     // [128][16]
  _Float16* s_spk = (_Float16*)(ws + OFF_SSPK);   // [2][128][1024]
  _Float16* r_spk = (_Float16*)(ws + OFF_RSPK);   // [2][128][2048]
  _Float16* o_spk = (_Float16*)(ws + OFF_OSPK);   // [2][128][32]
  unsigned* bar   = (unsigned*)(ws + OFF_BAR);
  const float*    sff   = (const float*)(ws + OFF_SFF);
  const _Float16* Wsr16 = (const _Float16*)(ws + OFF_WSR16);
  const _Float16* Wrd16 = (const _Float16*)(ws + OFF_WRD16);
  const _Float16* Wrr16 = (const _Float16*)(ws + OFF_WRR16);
  const _Float16* Wod16 = (const _Float16*)(ws + OFF_WOD16);
  const _Float16* Wor16 = (const _Float16*)(ws + OFF_WOR16);
  const float*    bodp  = (const float*)(ws + OFF_BODP);
  const float*    borp  = (const float*)(ws + OFF_BORP);

  float* out_s  = out;
  float* out_r  = out_s + (size_t)T_STEPS * Bz * S_SIZE;
  float* out_o  = out_r + (size_t)T_STEPS * Bz * R_SIZE;
  float* out_oV = out_o + (size_t)T_STEPS * Bz * 10;

  const int rl = lane & 15;          // N within tile (epilogue)
  const int mo = (lane >> 4) << 3;   // 0 or 8: M sub-block for C/D layout
  const int mt_blk = wave >> 6;      // uniform within a block (8 consecutive waves)

  for (int t = 0; t < T_STEPS; ++t) {
    const int wb = t & 1, rb = wb ^ 1;   // write / read spike buffer parity

    // ---------- stage 1: sensory layer (512 tiles, 1 per wave) ----------
    {
      const int mt = mt_blk, nt = wave & 63;
      // stage the shared s-spike panel (prev step) into LDS: 16 x 1024 f16
      panel_to_lds(s_spk + (size_t)rb * Bz * S_SIZE + (size_t)mt * 16 * S_SIZE,
                   ldsA, 16 * S_SIZE);
      const _Float16* Bw = Wsr16 + (size_t)nt * 16 * S_SIZE;
      v8f acc = {};
#pragma unroll 4
      for (int k = 0; k < S_SIZE; k += 32) {
        v16h a = load_frag(ldsA + k, S_SIZE, lane);        // ds_load_b128
        v16h b = load_frag(Bw + k, S_SIZE, lane);          // global_load_b128 (L2-hot)
        acc = wmma_f16(a, b, acc);
      }
      const int n = nt * 16 + rl;
      const int m0 = mt * 16 + mo;
      const float bias = bsr[n];
      _Float16* sw = s_spk + (size_t)wb * Bz * S_SIZE;
#pragma unroll
      for (int v = 0; v < 8; ++v) {
        const int m = m0 + v;
        const int idx = m * S_SIZE + n;
        const float memp = mem_s[idx];
        float base = 0.9f * memp + sff[idx]
                   + 0.1f * noise_s[(size_t)t * Bz * S_SIZE + idx] + acc[v] + bias;
        float mn  = (memp > 1.0f) ? 0.0f : base;      // zero-reset from prev mem
        float spk = (mn > 1.0f) ? 1.0f : 0.0f;
        mem_s[idx] = fmaxf(mn, 0.0f);                 // clamp carried into next step
        sw[idx] = (_Float16)spk;
        out_s[(size_t)t * Bz * S_SIZE + idx] = spk;
      }
    }
    device_barrier(bar);

    // ---------- stage 2: relay layer (1024 tiles, 1x2 N-pair per wave) ----------
    {
      const int mt = mt_blk;
      const int nt0 = (wave & 63) * 2;
      v8f acc0 = {}, acc1 = {};
      // phase A — dendrite: s_spk(current) @ Wrd.T   (K = 1024)
      panel_to_lds(s_spk + (size_t)wb * Bz * S_SIZE + (size_t)mt * 16 * S_SIZE,
                   ldsA, 16 * S_SIZE);
      const _Float16* B0 = Wrd16 + (size_t)nt0 * 16 * S_SIZE;
      const _Float16* B1 = B0 + 16 * S_SIZE;
#pragma unroll 2
      for (int k = 0; k < S_SIZE; k += 32) {
        v16h a  = load_frag(ldsA + k, S_SIZE, lane);
        v16h b0 = load_frag(B0 + k, S_SIZE, lane);
        v16h b1 = load_frag(B1 + k, S_SIZE, lane);
        acc0 = wmma_f16(a, b0, acc0);
        acc1 = wmma_f16(a, b1, acc1);
      }
      __syncthreads();                    // panel reads done before refill
      // phase B — recurrent: r_spk(prev) @ Wrr.T   (K = 2048)
      panel_to_lds(r_spk + (size_t)rb * Bz * R_SIZE + (size_t)mt * 16 * R_SIZE,
                   ldsA, 16 * R_SIZE);
      const _Float16* C0 = Wrr16 + (size_t)nt0 * 16 * R_SIZE;
      const _Float16* C1 = C0 + 16 * R_SIZE;
#pragma unroll 2
      for (int k = 0; k < R_SIZE; k += 32) {
        v16h a  = load_frag(ldsA + k, R_SIZE, lane);
        v16h b0 = load_frag(C0 + k, R_SIZE, lane);
        v16h b1 = load_frag(C1 + k, R_SIZE, lane);
        acc0 = wmma_f16(a, b0, acc0);
        acc1 = wmma_f16(a, b1, acc1);
      }
      _Float16* rw = r_spk + (size_t)wb * Bz * R_SIZE;
#pragma unroll
      for (int q = 0; q < 2; ++q) {
        v8f acc = q ? acc1 : acc0;
        const int n = (nt0 + q) * 16 + rl;
        const int m0 = mt * 16 + mo;
        const float bias = brd[n] + brr[n];
#pragma unroll
        for (int v = 0; v < 8; ++v) {
          const int m = m0 + v;
          const int idx = m * R_SIZE + n;
          const float memp = mem_r[idx];
          float base = 0.9f * memp + 0.1f * noise_r[(size_t)t * Bz * R_SIZE + idx]
                     + acc[v] + bias;
          float mn  = (memp > 1.0f) ? 0.0f : base;
          float spk = (mn > 1.0f) ? 1.0f : 0.0f;
          mem_r[idx] = fmaxf(mn, 0.0f);
          rw[idx] = (_Float16)spk;
          out_r[(size_t)t * Bz * R_SIZE + idx] = spk;
        }
      }
    }
    device_barrier(bar);

    // ---------- stage 3: output layer (8 M-tiles, N padded 10->16) ----------
    if (wave < 8) {
      const int mt = wave;
      v8f acc = {};
      const _Float16* A = r_spk + (size_t)wb * Bz * R_SIZE + (size_t)mt * 16 * R_SIZE;
#pragma unroll 4
      for (int k = 0; k < R_SIZE; k += 32) {
        v16h a = load_frag(A + k, R_SIZE, lane);
        v16h b = load_frag(Wod16 + k, R_SIZE, lane);
        acc = wmma_f16(a, b, acc);
      }
      {
        // recurrent: o_spk(prev)[128][32] @ Wor_p[16][32].T  (K padded 10->32)
        const _Float16* Ao = o_spk + (size_t)rb * Bz * 32 + mt * 16 * 32;
        v16h a = load_frag(Ao, 32, lane);
        v16h b = load_frag(Wor16, 32, lane);
        acc = wmma_f16(a, b, acc);
      }
      const int n = rl;                  // 0..15 (valid outputs: n < 10)
      const int m0 = mt * 16 + mo;
      const float bias = bodp[n] + borp[n];
      _Float16* ow = o_spk + (size_t)wb * Bz * 32;
#pragma unroll
      for (int v = 0; v < 8; ++v) {
        const int m = m0 + v;
        const float memp = mem_o[m * 16 + n];
        const float nz = (n < 10) ? noise_o[((size_t)t * Bz + m) * 10 + n] : 0.0f;
        float base = 0.9f * memp + 0.1f * nz + acc[v] + bias;
        float mn  = (memp > 1.0f) ? 0.0f : base;
        float spk = (mn > 1.0f) ? 1.0f : 0.0f;
        float mc  = fmaxf(mn, 0.0f);
        mem_o[m * 16 + n] = mc;
        ow[m * 32 + n] = (_Float16)spk;      // cols 16..31 stay zero from init
        if (n < 10) {
          out_o [((size_t)t * Bz + m) * 10 + n] = spk;
          out_oV[((size_t)t * Bz + m) * 10 + n] = mc;
        }
      }
    }
    device_barrier(bar);
  }
}

// ---------------- host entry ----------------
extern "C" void kernel_launch(void* const* d_in, const int* in_sizes, int n_in,
                              void* d_out, int out_size, void* d_ws, size_t ws_size,
                              hipStream_t stream) {
  (void)in_sizes; (void)n_in; (void)out_size; (void)ws_size;
  const float* stimulus = (const float*)d_in[0];
  const float* noise_s  = (const float*)d_in[1];
  const float* noise_r  = (const float*)d_in[2];
  const float* noise_o  = (const float*)d_in[3];
  const float* Wsd = (const float*)d_in[4];
  const float* bsd = (const float*)d_in[5];
  const float* Wrd = (const float*)d_in[6];
  const float* brd = (const float*)d_in[7];
  const float* Wod = (const float*)d_in[8];
  const float* bod = (const float*)d_in[9];
  const float* Wsr = (const float*)d_in[10];
  const float* bsr = (const float*)d_in[11];
  const float* Wrr = (const float*)d_in[12];
  const float* brr = (const float*)d_in[13];
  const float* Wor = (const float*)d_in[14];
  const float* bor = (const float*)d_in[15];
  char*  ws  = (char*)d_ws;
  float* out = (float*)d_out;

  // prologue: zero state + barrier, f16 weight copies, o-layer padding, sff GEMM
  k_zero<<<512, 256, 0, stream>>>((unsigned*)ws, (int)(OFF_ZERO_END / 4));
  k_cvt <<<1024, 256, 0, stream>>>(Wsr, (_Float16*)(ws + OFF_WSR16), S_SIZE * S_SIZE);
  k_cvt <<<2048, 256, 0, stream>>>(Wrd, (_Float16*)(ws + OFF_WRD16), R_SIZE * S_SIZE);
  k_cvt <<<4096, 256, 0, stream>>>(Wrr, (_Float16*)(ws + OFF_WRR16), R_SIZE * R_SIZE);
  k_pad_o<<<132, 256, 0, stream>>>(Wod, Wor, bod, bor, ws);
  k_sff<<<(Bz * S_SIZE + 255) / 256, 256, 0, stream>>>(stimulus, Wsd, bsd,
                                                       (float*)(ws + OFF_SFF));
  // persistent time-loop kernel: 150 dependent GEMM stages, device-wide barriers
  snn_persistent<<<NBLK, TPB, 0, stream>>>(noise_s, noise_r, noise_o,
                                           bsr, brd, brr, ws, out);
}
